// GraphSearchPolicy_64063732187322
// MI455X (gfx1250) — matrix operand order repaired and checked by reading, not compiled, not run on hardware
//
#include <hip/hip_runtime.h>
#include <hip/hip_bf16.h>

typedef __attribute__((ext_vector_type(16))) _Float16 v16h;
typedef __attribute__((ext_vector_type(8)))  _Float16 v8h;
typedef __attribute__((ext_vector_type(8)))  float    v8f;

#define B_SZ   1024
#define A_SZ   256
#define E_DIM  200
#define R_DIM  200
#define H_DIM  400
#define ACT_D  400
#define IN_D   800
#define N_REL  500
#define K2P    416   // ACT_D padded to 32
#define TP     224   // R_DIM padded to 32

// ---------------------------------------------------------------------------
// Prep kernels: f32 -> f16 converts with transpose + zero padding
// ---------------------------------------------------------------------------
__global__ void conv_wt(const float* __restrict__ W, _Float16* __restrict__ Wt,
                        int K, int N, int Kp) {
    int idx = blockIdx.x * 256 + threadIdx.x;
    if (idx >= N * Kp) return;
    int n = idx / Kp, k = idx % Kp;
    Wt[idx] = (k < K) ? (_Float16)W[(size_t)k * N + n] : (_Float16)0.f;
}

__global__ void conv_rows(const float* __restrict__ R, _Float16* __restrict__ Rh,
                          int rows, int K, int Kp) {
    int idx = blockIdx.x * 256 + threadIdx.x;
    if (idx >= rows * Kp) return;
    int n = idx / Kp, k = idx % Kp;
    Rh[idx] = (k < K) ? (_Float16)R[(size_t)n * K + k] : (_Float16)0.f;
}

// Gather E,H,Q into f16 X [B x 800]
__global__ void build_x(const int* __restrict__ e, const int* __restrict__ q,
                        const float* __restrict__ H,
                        const float* __restrict__ ent, const float* __restrict__ rel,
                        _Float16* __restrict__ Xh) {
    int idx = blockIdx.x * 256 + threadIdx.x;
    if (idx >= B_SZ * IN_D) return;
    int b = idx / IN_D, c = idx % IN_D;
    float v;
    if (c < E_DIM)            v = ent[(size_t)e[b] * E_DIM + c];
    else if (c < E_DIM+H_DIM) v = H[(size_t)b * H_DIM + (c - E_DIM)];
    else                      v = rel[(size_t)q[b] * R_DIM + (c - E_DIM - H_DIM)];
    Xh[idx] = (_Float16)v;
}

// ---------------------------------------------------------------------------
// Generic WMMA GEMM: C[M x N] = A[M x lda(f16)] * Bt[N x ldb(f16)]^T + bias
// A row-major (K padded in lda, pad zeroed), Bt pre-transposed N x Kpad.
// 256 threads = 8 waves, block tile 32x64 (2x4 waves of 16x16).
// ---------------------------------------------------------------------------
union FragU { v16h v; v8h h[2]; };

__global__ __launch_bounds__(256) void wmma_gemm(
    const _Float16* __restrict__ A, int lda,
    const _Float16* __restrict__ Bt, int ldb,
    const float* __restrict__ bias,
    float* __restrict__ Cf, int ldc,
    _Float16* __restrict__ Ch, int ldh,
    int M, int N, int relu)
{
    const int lane  = threadIdx.x & 31;
    const int wave  = threadIdx.x >> 5;
    const int l15   = lane & 15;
    const int hi    = lane >> 4;           // 0 or 1
    const int mBase = blockIdx.y * 32 + (wave >> 2) * 16;
    const int nBase = blockIdx.x * 64 + (wave & 3) * 16;

    // A fragment (16-bit 16x32 layout): lane<16 -> K {0..7,16..23}; lane>=16 -> K {8..15,24..31}
    const _Float16* aPtr = A + (size_t)(mBase + l15) * lda + hi * 8;
    // B fragment: column fixed per lane; lane<16 -> K {0..7,16..23}; lane>=16 -> K {8..15,24..31}
    const int bCol = min(nBase + l15, N - 1);
    const _Float16* bPtr = Bt + (size_t)bCol * ldb + hi * 8;

    v8f acc = {};
    for (int k0 = 0; k0 < lda; k0 += 32) {
        FragU af, bf;
        af.h[0] = *(const v8h*)(aPtr + k0);
        af.h[1] = *(const v8h*)(aPtr + k0 + 16);
        bf.h[0] = *(const v8h*)(bPtr + k0);
        bf.h[1] = *(const v8h*)(bPtr + k0 + 16);
        acc = __builtin_amdgcn_wmma_f32_16x16x32_f16(
            false, af.v, false, bf.v, (short)0, acc, false, false);
    }

    const int col = nBase + l15;
    const float bv = (bias && col < N) ? bias[col] : 0.f;
    #pragma unroll
    for (int r = 0; r < 8; ++r) {
        int row = mBase + r + hi * 8;
        float v = acc[r] + bv;
        if (relu) v = fmaxf(v, 0.f);
        if (Cf && col < N)  Cf[(size_t)row * ldc + col] = v;
        if (Ch && col < ldh)                     // zero-fills K padding region too
            Ch[(size_t)row * ldh + col] = (col < N) ? (_Float16)v : (_Float16)0.f;
    }
}

// ---------------------------------------------------------------------------
// In-place row softmax for relation attention logits [B x 500]
// ---------------------------------------------------------------------------
__global__ __launch_bounds__(512) void softmax_rel(float* __restrict__ L) {
    __shared__ float red[512];
    int b = blockIdx.x, t = threadIdx.x;
    float v = (t < N_REL) ? L[(size_t)b * N_REL + t] : -3.4e38f;
    red[t] = v; __syncthreads();
    for (int s = 256; s > 0; s >>= 1) { if (t < s) red[t] = fmaxf(red[t], red[t + s]); __syncthreads(); }
    float mx = red[0]; __syncthreads();
    float ex = (t < N_REL) ? __expf(v - mx) : 0.f;
    red[t] = ex; __syncthreads();
    for (int s = 256; s > 0; s >>= 1) { if (t < s) red[t] += red[t + s]; __syncthreads(); }
    float sum = red[0];
    if (t < N_REL) L[(size_t)b * N_REL + t] = ex / sum;
}

// ---------------------------------------------------------------------------
// Fused action scores: gather rel/ent embeddings, dot with X2 (in LDS),
// mask, softmax over A=256, entropy. One block per batch row.
// ---------------------------------------------------------------------------
__global__ __launch_bounds__(256) void scores_softmax(
    const int* __restrict__ r_space, const int* __restrict__ e_space,
    const int* __restrict__ mask,
    const float* __restrict__ rel_emb, const float* __restrict__ ent_emb,
    const float* __restrict__ X2,          // [B x 400] f32
    float* __restrict__ dist_out,          // [B x 256]
    float* __restrict__ ent_out)           // [B]
{
    __shared__ float sX[ACT_D];
    __shared__ float red[256];
    const int b = blockIdx.x, a = threadIdx.x;
    for (int i = a; i < ACT_D; i += 256) sX[i] = X2[(size_t)b * ACT_D + i];
    __syncthreads();

    const int ri = r_space[(size_t)b * A_SZ + a];
    const int ei = e_space[(size_t)b * A_SZ + a];
    const float4* rp = (const float4*)(rel_emb + (size_t)ri * R_DIM);
    const float4* ep = (const float4*)(ent_emb + (size_t)ei * E_DIM);
    const float4* x0 = (const float4*)sX;
    const float4* x1 = (const float4*)(sX + R_DIM);
    float accum = 0.f;
    #pragma unroll 5
    for (int i = 0; i < R_DIM / 4; ++i) {
        float4 rv = rp[i], xv = x0[i];
        accum += rv.x * xv.x + rv.y * xv.y + rv.z * xv.z + rv.w * xv.w;
        float4 ev = ep[i], yv = x1[i];
        accum += ev.x * yv.x + ev.y * yv.y + ev.z * yv.z + ev.w * yv.w;
    }
    const int m = mask[(size_t)b * A_SZ + a];
    float score = accum - (1.f - (float)m) * 1e31f;

    // max
    red[a] = score; __syncthreads();
    for (int s = 128; s > 0; s >>= 1) { if (a < s) red[a] = fmaxf(red[a], red[a + s]); __syncthreads(); }
    float mx = red[0]; __syncthreads();
    // sum of exp
    float ex = __expf(score - mx);
    red[a] = ex; __syncthreads();
    for (int s = 128; s > 0; s >>= 1) { if (a < s) red[a] += red[a + s]; __syncthreads(); }
    float sum = red[0]; __syncthreads();
    float d = ex / sum;
    dist_out[(size_t)b * A_SZ + a] = d;
    // entropy
    red[a] = -d * __logf(d + 1e-20f); __syncthreads();
    for (int s = 128; s > 0; s >>= 1) { if (a < s) red[a] += red[a + s]; __syncthreads(); }
    if (a == 0) ent_out[b] = red[0];
}

// ---------------------------------------------------------------------------
static inline size_t align256(size_t x) { return (x + 255) & ~(size_t)255; }

extern "C" void kernel_launch(void* const* d_in, const int* in_sizes, int n_in,
                              void* d_out, int out_size, void* d_ws, size_t ws_size,
                              hipStream_t stream) {
    const int*   e        = (const int*)  d_in[0];
    const int*   q        = (const int*)  d_in[1];
    const float* H        = (const float*)d_in[2];
    const int*   r_space  = (const int*)  d_in[3];
    const int*   e_space  = (const int*)  d_in[4];
    const int*   amask    = (const int*)  d_in[5];
    const float* ent_emb  = (const float*)d_in[6];
    const float* rel_emb  = (const float*)d_in[7];
    const float* W1       = (const float*)d_in[8];
    const float* b1       = (const float*)d_in[9];
    const float* W2       = (const float*)d_in[10];
    const float* b2       = (const float*)d_in[11];
    const float* W_att    = (const float*)d_in[12];
    const float* b_att    = (const float*)d_in[13];

    float* out      = (float*)d_out;
    float* dist_out = out;                                   // [1024 x 256]
    float* ent_out  = out + (size_t)B_SZ * A_SZ;             // [1024]
    float* relAtt   = ent_out + B_SZ;                        // [1024 x 500]

    // workspace carve-up
    char* ws = (char*)d_ws;
    size_t off = 0;
    _Float16* Xh    = (_Float16*)(ws + off); off = align256(off + (size_t)B_SZ * IN_D  * 2);
    _Float16* W1t   = (_Float16*)(ws + off); off = align256(off + (size_t)ACT_D * IN_D * 2);
    _Float16* X1h   = (_Float16*)(ws + off); off = align256(off + (size_t)B_SZ * K2P   * 2);
    _Float16* W2t   = (_Float16*)(ws + off); off = align256(off + (size_t)ACT_D * K2P  * 2);
    float*    X2f   = (float*)   (ws + off); off = align256(off + (size_t)B_SZ * ACT_D * 4);
    _Float16* X2h   = (_Float16*)(ws + off); off = align256(off + (size_t)B_SZ * K2P   * 2);
    _Float16* Watt  = (_Float16*)(ws + off); off = align256(off + (size_t)R_DIM * K2P  * 2);
    _Float16* Th    = (_Float16*)(ws + off); off = align256(off + (size_t)B_SZ * TP    * 2);
    _Float16* relh  = (_Float16*)(ws + off); off = align256(off + (size_t)N_REL * TP   * 2);
    (void)off; (void)ws_size; (void)in_sizes; (void)n_in; (void)out_size;

    // --- prep ---
    {
        int n = ACT_D * IN_D;                    // W1t 400x800
        conv_wt<<<(n + 255) / 256, 256, 0, stream>>>(W1, W1t, IN_D, ACT_D, IN_D);
        n = ACT_D * K2P;                         // W2t 400x416
        conv_wt<<<(n + 255) / 256, 256, 0, stream>>>(W2, W2t, ACT_D, ACT_D, K2P);
        n = R_DIM * K2P;                         // Watt 200x416
        conv_wt<<<(n + 255) / 256, 256, 0, stream>>>(W_att, Watt, ACT_D, R_DIM, K2P);
        n = N_REL * TP;                          // relh 500x224 (already "N x K" layout)
        conv_rows<<<(n + 255) / 256, 256, 0, stream>>>(rel_emb, relh, N_REL, R_DIM, TP);
        n = B_SZ * IN_D;
        build_x<<<(n + 255) / 256, 256, 0, stream>>>(e, q, H, ent_emb, rel_emb, Xh);
    }

    // --- GEMM 1: X1 = relu(X @ W1 + b1)  [1024x800 x 800x400] ---
    wmma_gemm<<<dim3((ACT_D + 63) / 64, B_SZ / 32), 256, 0, stream>>>(
        Xh, IN_D, W1t, IN_D, b1, nullptr, 0, X1h, K2P, B_SZ, ACT_D, 1);

    // --- GEMM 2: X2 = X1 @ W2 + b2 ---
    wmma_gemm<<<dim3((ACT_D + 63) / 64, B_SZ / 32), 256, 0, stream>>>(
        X1h, K2P, W2t, K2P, b2, X2f, ACT_D, X2h, K2P, B_SZ, ACT_D, 0);

    // --- GEMM 3: T = X2 @ W_att + b_att  [1024x400 x 400x200] ---
    wmma_gemm<<<dim3((R_DIM + 63) / 64, B_SZ / 32), 256, 0, stream>>>(
        X2h, K2P, Watt, K2P, b_att, nullptr, 0, Th, TP, B_SZ, R_DIM, 0);

    // --- GEMM 4: relation logits = T @ rel_emb^T  [1024x200 x 200x500] ---
    wmma_gemm<<<dim3((N_REL + 63) / 64, B_SZ / 32), 256, 0, stream>>>(
        Th, TP, relh, TP, nullptr, relAtt, N_REL, nullptr, 0, B_SZ, N_REL, 0);

    // --- relation softmax (in place on output) ---
    softmax_rel<<<B_SZ, 512, 0, stream>>>(relAtt);

    // --- fused action scores + masked softmax + entropy ---
    scores_softmax<<<B_SZ, 256, 0, stream>>>(
        r_space, e_space, amask, rel_emb, ent_emb, X2f, dist_out, ent_out);
}
